// EventResidualInjector_27779848470665
// MI455X (gfx1250) — matrix operand order repaired and checked by reading, not compiled, not run on hardware
//
#include <hip/hip_runtime.h>
#include <hip/hip_bf16.h>
#include <math.h>

typedef __attribute__((ext_vector_type(16))) _Float16 v16h;
typedef __attribute__((ext_vector_type(8)))  float    v8f;

#define NN   500
#define NP   512
#define DDIM 64
#define NE   2000
#define NE2  2500
#define SDIM 288
#define FLAT 2016

// ---- d_out offsets (floats), tuple order of reference ----
#define OFF_SEED  0L
#define OFF_DIF   32000L
#define OFF_DELTA 64000L
#define OFF_HF    64576000L          // 64000 + 2016*500*64
#define OFF_PRED  129088000L
#define OFF_MASK  130096000L

// ---- workspace byte offsets ----
#define WS_ACCA   0                  // 512*64 f32 (GCN accumulation)
#define WS_EMB    131072             // 512*64 f32
#define WS_XF     262144             // 512*64 f32 (GCN out, relu)
#define WS_XW     393216             // 512*64 f32 (xw scratch)
#define WS_DIF    524288             // 512*64 f32 (diffused)
#define WS_SEEDF  655360             // 512*64 f32 (delta_seed)
#define WS_DEG    786432             // 512 f32
#define WS_DINV   788480             // 512 f32
#define WS_AS     790528             // 512 f32
#define WS_AD     792576             // 512 f32
#define WS_RHO    794624             // 512 f32
#define WS_DIF16  796672             // 512*64 f16
#define WS_SEED16 862208             // 512*64 f16
#define WS_TIME16 927744             // 2016*64 f16
#define WS_W1F    1185792            // 16384 f16 (fg1 frag-major)
#define WS_W2F    1218560            // 4096 f16  (fg2 frag-major)
#define WS_S1F    1226752            // 2048 f16  (sp1 frag-major)

// =============== WMMA helpers (wave32, gfx1250 layouts) ===============
__device__ __forceinline__ v8f wmma_f16(v16h a, v16h b, v8f c) {
  // D = A(16x32 f16) * B(32x16 f16) + C(16x16 f32)
  return __builtin_amdgcn_wmma_f32_16x16x32_f16(false, a, false, b, (short)0, c, false, false);
}

// 16-bit A 16x32 layout: lane l=L&15 is row M; VGPR j covers K = (j<4?0:16)+hi*8+2*(j&3)+{0,1}
__device__ __forceinline__ v16h a_from_f16(const _Float16* row, int kb, int hi) {
  union { v16h v; unsigned u[8]; } x;
#pragma unroll
  for (int j = 0; j < 8; ++j) {
    int k = kb + ((j & 4) ? 16 : 0) + hi * 8 + ((j & 3) << 1);
    x.u[j] = *(const unsigned*)(row + k);     // two consecutive halves
  }
  return x.v;
}
__device__ __forceinline__ v16h a_from_f32(const float* row, int kb, int hi) {
  union { v16h v; _Float16 h[16]; } x;
#pragma unroll
  for (int j = 0; j < 8; ++j) {
    int k = kb + ((j & 4) ? 16 : 0) + hi * 8 + ((j & 3) << 1);
    float2 f2 = *(const float2*)(row + k);
    x.h[2 * j]     = (_Float16)f2.x;
    x.h[2 * j + 1] = (_Float16)f2.y;
  }
  return x.v;
}

__device__ __forceinline__ float sigmf(float x) { return 1.f / (1.f + expf(-x)); }

// =============== prologue kernels ===============
__global__ void k_init(float* accA, float* deg) {
  int idx = blockIdx.x * blockDim.x + threadIdx.x;
  if (idx < NP * DDIM) accA[idx] = 0.f;
  if (idx < NP) deg[idx] = 0.f;
}

__global__ void k_deg(const int* __restrict__ ei, float* deg) {
  int e = blockIdx.x * blockDim.x + threadIdx.x;
  if (e < NE) atomicAdd(&deg[ei[NE + e]], 1.f);
}

__global__ void k_dinv(const float* deg, float* dinv) {
  int n = blockIdx.x * blockDim.x + threadIdx.x;
  if (n < NP) dinv[n] = (n < NN) ? rsqrtf(deg[n] + 1.f) : 0.f;  // +1 self loop
}

__global__ void k_emb(const float* __restrict__ ev, const float* __restrict__ W1,
                      const float* __restrict__ b1, const float* __restrict__ W2,
                      const float* __restrict__ b2, float* __restrict__ emb) {
  int n = blockIdx.x, tid = threadIdx.x;
  __shared__ float se[8], sh[64];
  if (tid < 8) se[tid] = ev[n * 8 + tid];
  __syncthreads();
  float h = b1[tid];
#pragma unroll
  for (int k = 0; k < 8; ++k) h += se[k] * W1[k * 64 + tid];
  sh[tid] = fmaxf(h, 0.f);
  __syncthreads();
  float o = b2[tid];
#pragma unroll 8
  for (int k = 0; k < 64; ++k) o += sh[k] * W2[k * 64 + tid];
  emb[n * 64 + tid] = o;
}

__global__ void k_node_mm64(const float* __restrict__ X, const float* __restrict__ W,
                            float* __restrict__ Y) {
  int n = blockIdx.x, tid = threadIdx.x;
  __shared__ float sx[64];
  sx[tid] = X[n * 64 + tid];
  __syncthreads();
  float a = 0.f;
#pragma unroll 8
  for (int k = 0; k < 64; ++k) a += sx[k] * W[k * 64 + tid];
  Y[n * 64 + tid] = a;
}

__global__ void k_scatter_gcn(const int* __restrict__ ei, const float* __restrict__ dinv,
                              const float* __restrict__ xw, float* accA) {
  int e = blockIdx.x, tid = threadIdx.x;
  int s, d2;
  if (e < NE) { s = ei[e]; d2 = ei[NE + e]; } else { s = e - NE; d2 = s; }
  float nm = dinv[s] * dinv[d2];
  atomicAdd(&accA[d2 * 64 + tid], nm * xw[s * 64 + tid]);
}

__global__ void k_bias_relu(const float* __restrict__ accA, const float* __restrict__ b,
                            float* __restrict__ xf) {
  int n = blockIdx.x, tid = threadIdx.x;
  xf[n * 64 + tid] = (n < NN) ? fmaxf(accA[n * 64 + tid] + b[tid], 0.f) : 0.f;
}

__global__ void k_attdot(const float* __restrict__ xw, const float* __restrict__ attS,
                         const float* __restrict__ attD, float* a_s, float* a_d) {
  int n = blockIdx.x, tid = threadIdx.x;
  __shared__ float r1[64], r2[64];
  float x = xw[n * 64 + tid];
  r1[tid] = x * attS[tid];
  r2[tid] = x * attD[tid];
  __syncthreads();
  for (int s2 = 32; s2 > 0; s2 >>= 1) {
    if (tid < s2) { r1[tid] += r1[tid + s2]; r2[tid] += r2[tid + s2]; }
    __syncthreads();
  }
  if (tid == 0) { a_s[n] = r1[0]; a_d[n] = r2[0]; }
}

__global__ void k_gat(const int* __restrict__ ei, const float* __restrict__ xw,
                      const float* __restrict__ a_s, const float* __restrict__ a_d,
                      const float* __restrict__ gat_b, float* __restrict__ dif,
                      _Float16* __restrict__ dif16) {
  int n = blockIdx.x, tid = threadIdx.x;
  __shared__ float red[64];
  if (n >= NN) { dif[n * 64 + tid] = 0.f; dif16[n * 64 + tid] = (_Float16)0.f; return; }
  const int* src = ei;
  const int* dst = ei + NE;
  float adn = a_d[n];
  // pass 1: segment max (self loop guarantees >=1 member)
  float mx = -1e30f;
  for (int e = tid; e < NE2; e += 64) {
    int s, d2;
    if (e < NE) { s = src[e]; d2 = dst[e]; } else { s = e - NE; d2 = s; }
    if (d2 == n) {
      float v = a_s[s] + adn;
      if (v < 0.f) v *= 0.2f;
      mx = fmaxf(mx, v);
    }
  }
  red[tid] = mx; __syncthreads();
  for (int s2 = 32; s2 > 0; s2 >>= 1) { if (tid < s2) red[tid] = fmaxf(red[tid], red[tid + s2]); __syncthreads(); }
  float M = red[0]; __syncthreads();
  // pass 2: sum of exp
  float sm = 0.f;
  for (int e = tid; e < NE2; e += 64) {
    int s, d2;
    if (e < NE) { s = src[e]; d2 = dst[e]; } else { s = e - NE; d2 = s; }
    if (d2 == n) {
      float v = a_s[s] + adn;
      if (v < 0.f) v *= 0.2f;
      sm += expf(v - M);
    }
  }
  red[tid] = sm; __syncthreads();
  for (int s2 = 32; s2 > 0; s2 >>= 1) { if (tid < s2) red[tid] += red[tid + s2]; __syncthreads(); }
  float S = red[0]; __syncthreads();
  // pass 3: weighted aggregation, thread owns dim
  float acc = 0.f;
  for (int e = 0; e < NE2; ++e) {
    int s, d2;
    if (e < NE) { s = src[e]; d2 = dst[e]; } else { s = e - NE; d2 = s; }
    if (d2 == n) {
      float v = a_s[s] + adn;
      if (v < 0.f) v *= 0.2f;
      float al = expf(v - M) / S;
      acc += al * xw[s * 64 + tid];
    }
  }
  float o = fmaxf(acc + gat_b[tid], 0.f);
  dif[n * 64 + tid] = o;
  dif16[n * 64 + tid] = (_Float16)o;
}

__global__ void k_gate(const float* __restrict__ H, const int* __restrict__ wd,
                       const int* __restrict__ sl, const float* __restrict__ dif,
                       const float* __restrict__ Wg, const float* __restrict__ bg,
                       const float* __restrict__ Wr1, const float* __restrict__ br1,
                       const float* __restrict__ Wr2, const float* __restrict__ br2,
                       const float* __restrict__ Wd1, const float* __restrict__ bd1,
                       const float* __restrict__ Wd2, const float* __restrict__ bd2,
                       float* __restrict__ seedf, _Float16* __restrict__ seed16,
                       float* __restrict__ rho, float* __restrict__ out) {
  int n = blockIdx.x, tid = threadIdx.x;
  __shared__ float sIn[192];
  __shared__ float sh[64];
  __shared__ float sr[64];
  if (n >= NN) {
    seedf[n * 64 + tid] = 0.f;
    seed16[n * 64 + tid] = (_Float16)0.f;
    if (tid == 0) rho[n] = 1.f;
    return;
  }
  int start = wd[0] * SDIM + sl[0];
  const float* H0 = H + ((size_t)start * NN + n) * 64;
  sIn[tid] = H0[tid];
  sIn[64 + tid] = dif[n * 64 + tid];
  __syncthreads();
  float gacc = bg[tid], hacc = br1[tid];
#pragma unroll 4
  for (int k = 0; k < 128; ++k) {
    float xv = sIn[k];
    gacc += xv * Wg[k * 64 + tid];
    hacc += xv * Wr1[k * 64 + tid];
  }
  float gate = sigmf(gacc);
  sh[tid] = fmaxf(hacc, 0.f);
  __syncthreads();
  float dr = br2[tid];
#pragma unroll 8
  for (int k = 0; k < 64; ++k) dr += sh[k] * Wr2[k * 64 + tid];
  float seed = gate * dr;
  seedf[n * 64 + tid] = seed;
  seed16[n * 64 + tid] = (_Float16)seed;
  out[OFF_SEED + n * 64 + tid] = seed;
  out[OFF_DIF + n * 64 + tid] = sIn[64 + tid];
  sIn[128 + tid] = seed;
  __syncthreads();
  float h2 = bd1[tid];
#pragma unroll 4
  for (int k = 0; k < 192; ++k) h2 += sIn[k] * Wd1[k * 64 + tid];
  sh[tid] = fmaxf(h2, 0.f);
  __syncthreads();
  sr[tid] = sh[tid] * Wd2[tid];
  __syncthreads();
  for (int s2 = 32; s2 > 0; s2 >>= 1) { if (tid < s2) sr[tid] += sr[tid + s2]; __syncthreads(); }
  if (tid == 0) {
    float r = sr[0] + bd2[0];
    float spv = (r > 20.f) ? r : log1pf(expf(r));
    rho[n] = spv + 1e-4f;
  }
}

__global__ void k_time(const float* __restrict__ Wt1, const float* __restrict__ bt1,
                       const float* __restrict__ Wt2, const float* __restrict__ bt2,
                       _Float16* __restrict__ time16) {
  int t = blockIdx.x, tid = threadIdx.x;
  __shared__ float sh[64];
  float day = (float)t * (1.f / 288.f);
  float wk  = (float)t * (1.f / 2016.f);
  float da = 6.2831853071795864f * day;
  float wa = 6.2831853071795864f * wk;
  float f[6] = {day, wk, sinf(da), cosf(da), sinf(wa), cosf(wa)};
  float h = bt1[tid];
#pragma unroll
  for (int k = 0; k < 6; ++k) h += f[k] * Wt1[k * 64 + tid];
  sh[tid] = fmaxf(h, 0.f);
  __syncthreads();
  float o = bt2[tid];
#pragma unroll 8
  for (int k = 0; k < 64; ++k) o += sh[k] * Wt2[k * 64 + tid];
  time16[t * 64 + tid] = (_Float16)o;
}

// Re-lay weights into frag-major f16 B-layout: frag f=(kc*nnt+nt), lane, i -> W[kc*32+((i&8)<<1)+hi*8+(i&7)][nt*16+l]
__global__ void k_wfrag(const float* __restrict__ W, _Float16* __restrict__ out, int K, int N) {
  int idx = blockIdx.x * blockDim.x + threadIdx.x;
  int total = (K >> 5) * (N >> 4) * 512;
  if (idx >= total) return;
  int i = idx & 15;
  int lane = (idx >> 4) & 31;
  int f = idx >> 9;
  int nnt = N >> 4;
  int kc = f / nnt, nt = f - kc * nnt;
  int l = lane & 15, hh = lane >> 4;
  out[idx] = (_Float16)W[(kc * 32 + ((i & 8) << 1) + hh * 8 + (i & 7)) * N + (nt * 16 + l)];
}

// =============== the monster kernel ===============
__global__ __launch_bounds__(32) void k_main(
    const float* __restrict__ H, const int* __restrict__ wd, const int* __restrict__ sl,
    const _Float16* __restrict__ dif16, const _Float16* __restrict__ seed16,
    const _Float16* __restrict__ time16, const float* __restrict__ seedf,
    const float* __restrict__ rho,
    const _Float16* __restrict__ W1f, const _Float16* __restrict__ W2f,
    const _Float16* __restrict__ S1f,
    const float* __restrict__ b1, const float* __restrict__ b2,
    const float* __restrict__ bs1, const float* __restrict__ sp2w,
    const float* __restrict__ bs2, float* __restrict__ out) {
  __shared__ _Float16 lds_hid[16 * 64];
  __shared__ _Float16 lds_hf[16 * 64];
  __shared__ float lds_r[16 * 32];

  const int tid = threadIdx.x;
  const int l = tid & 15, hi = tid >> 4;
  const int s = blockIdx.y;
  const int n0 = blockIdx.x * 16;
  const int start = wd[0] * SDIM + sl[0];

  float* outD = out + OFF_DELTA;
  float* outH = out + OFF_HF;
  float* outP = out + OFF_PRED;
  float* outM = out + OFF_MASK;
  const size_t slot_base = (size_t)s * NN * DDIM;

  if (s >= start) {
    const int t_rel = s - start;
    // ---- fg1: fusion(16x256) @ W1(256x64), f16 WMMA, fp32 accumulate ----
    v8f acc[4] = {};
    const v16h* W1v = (const v16h*)W1f;
#pragma unroll
    for (int kc = 0; kc < 8; ++kc) {
      const int reg = kc >> 1;
      const int kb = (kc & 1) * 32;
      const int node = n0 + l;
      v16h a = {};
      if (reg == 0) {
        if (node < NN) a = a_from_f32(H + slot_base + (size_t)node * DDIM, kb, hi);
      } else if (reg == 1) {
        a = a_from_f16(dif16 + node * DDIM, kb, hi);
      } else if (reg == 2) {
        a = a_from_f16(seed16 + node * DDIM, kb, hi);
      } else {
        a = a_from_f16(time16 + (size_t)t_rel * DDIM, kb, hi);
      }
#pragma unroll
      for (int nt = 0; nt < 4; ++nt) {
        v16h b = W1v[(kc * 4 + nt) * 32 + tid];
        acc[nt] = wmma_f16(a, b, acc[nt]);
      }
    }
    // bias + relu -> LDS (D layout: row M = v+8*hi, col = nt*16+l)
#pragma unroll
    for (int nt = 0; nt < 4; ++nt)
#pragma unroll
      for (int v = 0; v < 8; ++v) {
        float x = acc[nt][v] + b1[nt * 16 + l];
        lds_hid[(v + 8 * hi) * DDIM + nt * 16 + l] = (_Float16)fmaxf(x, 0.f);
      }
    __syncthreads();
    // ---- fg2: hidden(16x64) @ W2(64x64) ----
    v8f g[4] = {};
    const v16h* W2v = (const v16h*)W2f;
#pragma unroll
    for (int kc = 0; kc < 2; ++kc) {
      v16h a = a_from_f16(lds_hid + l * DDIM, kc * 32, hi);
#pragma unroll
      for (int nt = 0; nt < 4; ++nt) {
        v16h b = W2v[(kc * 4 + nt) * 32 + tid];
        g[nt] = wmma_f16(a, b, g[nt]);
      }
    }
    // ---- delta injection + H_final ----
    const float dd = (float)t_rel * (1.0f / 288.0f);
#pragma unroll
    for (int nt = 0; nt < 4; ++nt)
#pragma unroll
      for (int v = 0; v < 8; ++v) {
        int m = v + 8 * hi;
        int node = n0 + m;
        int d = nt * 16 + l;
        float hfv = 0.f;
        if (node < NN) {
          float gf = sigmf(g[nt][v] + b2[d]);
          float alpha = expf(-dd * rho[node]);
          float de = alpha * gf * seedf[node * DDIM + d];
          size_t off = slot_base + (size_t)node * DDIM + d;
          float hv = H[off];
          hfv = hv + de;
          outD[off] = de;
          outH[off] = hfv;
        }
        lds_hf[m * DDIM + d] = (_Float16)hfv;
      }
    __syncthreads();
  } else {
    // past slots: delta = 0, H_final = H
    for (int idx = tid; idx < 16 * DDIM; idx += 32) {
      int m = idx >> 6, d = idx & 63;
      int node = n0 + m;
      float hv = 0.f;
      if (node < NN) {
        size_t off = slot_base + (size_t)node * DDIM + d;
        hv = H[off];
        outD[off] = 0.f;
        outH[off] = hv;
      }
      lds_hf[idx] = (_Float16)hv;
    }
    __syncthreads();
  }

  // ---- speed head: relu(HF(16x64) @ sp1(64x32) + b) @ sp2(32x1) + b ----
  v8f p[2] = {};
  const v16h* S1v = (const v16h*)S1f;
#pragma unroll
  for (int kc = 0; kc < 2; ++kc) {
    v16h a = a_from_f16(lds_hf + l * DDIM, kc * 32, hi);
#pragma unroll
    for (int nt = 0; nt < 2; ++nt) {
      v16h b = S1v[(kc * 2 + nt) * 32 + tid];
      p[nt] = wmma_f16(a, b, p[nt]);
    }
  }
#pragma unroll
  for (int nt = 0; nt < 2; ++nt)
#pragma unroll
    for (int v = 0; v < 8; ++v)
      lds_r[(v + 8 * hi) * 32 + nt * 16 + l] = fmaxf(p[nt][v] + bs1[nt * 16 + l], 0.f);
  __syncthreads();
  if (hi == 0) {
    int node = n0 + l;
    if (node < NN) {
      float a2 = bs2[0];
#pragma unroll 8
      for (int k = 0; k < 32; ++k) a2 += lds_r[l * 32 + k] * sp2w[k];
      size_t po = (size_t)s * NN + node;
      outP[po] = a2;
      outM[po] = (s >= start) ? 1.0f : 0.0f;
    }
  }
}

// =============== launcher ===============
extern "C" void kernel_launch(void* const* d_in, const int* in_sizes, int n_in,
                              void* d_out, int out_size, void* d_ws, size_t ws_size,
                              hipStream_t stream) {
  const float* H    = (const float*)d_in[0];
  const float* ev   = (const float*)d_in[1];
  const int*   ei   = (const int*)d_in[2];
  const int*   wd   = (const int*)d_in[3];
  const int*   sl   = (const int*)d_in[4];
  const float* enc1W = (const float*)d_in[5];  const float* enc1b = (const float*)d_in[6];
  const float* enc2W = (const float*)d_in[7];  const float* enc2b = (const float*)d_in[8];
  const float* gcnW  = (const float*)d_in[9];  const float* gcnb  = (const float*)d_in[10];
  const float* gatW  = (const float*)d_in[11];
  const float* attS  = (const float*)d_in[12]; const float* attD  = (const float*)d_in[13];
  const float* gatb  = (const float*)d_in[14];
  const float* gateW = (const float*)d_in[15]; const float* gateb = (const float*)d_in[16];
  const float* r1W   = (const float*)d_in[17]; const float* r1b   = (const float*)d_in[18];
  const float* r2W   = (const float*)d_in[19]; const float* r2b   = (const float*)d_in[20];
  const float* t1W   = (const float*)d_in[21]; const float* t1b   = (const float*)d_in[22];
  const float* t2W   = (const float*)d_in[23]; const float* t2b   = (const float*)d_in[24];
  const float* dc1W  = (const float*)d_in[25]; const float* dc1b  = (const float*)d_in[26];
  const float* dc2W  = (const float*)d_in[27]; const float* dc2b  = (const float*)d_in[28];
  const float* fg1W  = (const float*)d_in[29]; const float* fg1b  = (const float*)d_in[30];
  const float* fg2W  = (const float*)d_in[31]; const float* fg2b  = (const float*)d_in[32];
  const float* sp1W  = (const float*)d_in[33]; const float* sp1b  = (const float*)d_in[34];
  const float* sp2W  = (const float*)d_in[35]; const float* sp2b  = (const float*)d_in[36];

  float* out = (float*)d_out;
  char* ws = (char*)d_ws;
  float* accA  = (float*)(ws + WS_ACCA);
  float* emb   = (float*)(ws + WS_EMB);
  float* xf    = (float*)(ws + WS_XF);
  float* xw    = (float*)(ws + WS_XW);
  float* dif   = (float*)(ws + WS_DIF);
  float* seedf = (float*)(ws + WS_SEEDF);
  float* deg   = (float*)(ws + WS_DEG);
  float* dinv  = (float*)(ws + WS_DINV);
  float* a_s   = (float*)(ws + WS_AS);
  float* a_d   = (float*)(ws + WS_AD);
  float* rho   = (float*)(ws + WS_RHO);
  _Float16* dif16  = (_Float16*)(ws + WS_DIF16);
  _Float16* seed16 = (_Float16*)(ws + WS_SEED16);
  _Float16* time16 = (_Float16*)(ws + WS_TIME16);
  _Float16* W1f = (_Float16*)(ws + WS_W1F);
  _Float16* W2f = (_Float16*)(ws + WS_W2F);
  _Float16* S1f = (_Float16*)(ws + WS_S1F);

  k_init<<<dim3(131), dim3(256), 0, stream>>>(accA, deg);
  k_deg<<<dim3(8), dim3(256), 0, stream>>>(ei, deg);
  k_dinv<<<dim3(2), dim3(256), 0, stream>>>(deg, dinv);
  k_emb<<<dim3(500), dim3(64), 0, stream>>>(ev, enc1W, enc1b, enc2W, enc2b, emb);
  k_node_mm64<<<dim3(500), dim3(64), 0, stream>>>(emb, gcnW, xw);
  k_scatter_gcn<<<dim3(2500), dim3(64), 0, stream>>>(ei, dinv, xw, accA);
  k_bias_relu<<<dim3(512), dim3(64), 0, stream>>>(accA, gcnb, xf);
  k_node_mm64<<<dim3(500), dim3(64), 0, stream>>>(xf, gatW, xw);
  k_attdot<<<dim3(500), dim3(64), 0, stream>>>(xw, attS, attD, a_s, a_d);
  k_gat<<<dim3(512), dim3(64), 0, stream>>>(ei, xw, a_s, a_d, gatb, dif, dif16);
  k_gate<<<dim3(512), dim3(64), 0, stream>>>(H, wd, sl, dif, gateW, gateb, r1W, r1b, r2W, r2b,
                                             dc1W, dc1b, dc2W, dc2b, seedf, seed16, rho, out);
  k_time<<<dim3(2016), dim3(64), 0, stream>>>(t1W, t1b, t2W, t2b, time16);
  k_wfrag<<<dim3(64), dim3(256), 0, stream>>>(fg1W, W1f, 256, 64);
  k_wfrag<<<dim3(16), dim3(256), 0, stream>>>(fg2W, W2f, 64, 64);
  k_wfrag<<<dim3(8), dim3(256), 0, stream>>>(sp1W, S1f, 64, 32);
  k_main<<<dim3(32, 2016), dim3(32), 0, stream>>>(H, wd, sl, dif16, seed16, time16, seedf, rho,
                                                  W1f, W2f, S1f, fg1b, fg2b, sp1b, sp2W, sp2b, out);
}
